// FusedAttention_51659866636458
// MI455X (gfx1250) — matrix operand order repaired and checked by reading, not compiled, not run on hardware
//
#include <hip/hip_runtime.h>
#include <hip/hip_bf16.h>

// ---------------------------------------------------------------------------
// Fused llama attention prefill for MI455X (gfx1250), wave32 + WMMA f16.
//   x:(2,2048,2048) f16 ; wq:(2048,2048) wk/wv:(2048,512) wo:(2048,2048)
//   out = (softmax(rope(xWq) rope(xWk)^T / 8 + causal) * (xWv)) Wo
// Strategy: pre-transpose weights/V once (amortized 32x), then all hot-loop
// LDS staging is 16B-aligned b128 async-to-LDS (ASYNCcnt), all WMMA
// fragments are paired ds_load_b128.
// ---------------------------------------------------------------------------

#define Bdim   2
#define Sdim   2048
#define Hdim   2048
#define NH     32
#define NKV    8
#define HD     64

typedef _Float16 h8  __attribute__((ext_vector_type(8)));
typedef _Float16 h16 __attribute__((ext_vector_type(16)));
typedef float    f8  __attribute__((ext_vector_type(8)));
typedef int      i4v __attribute__((vector_size(16)));   // matches builtin param

__device__ __forceinline__ f8 f8zero() {
    f8 z = {0.f, 0.f, 0.f, 0.f, 0.f, 0.f, 0.f, 0.f};
    return z;
}

// Build a 16-half WMMA fragment from two contiguous 16B LDS chunks.
__device__ __forceinline__ h16 ld_frag(const _Float16* p0, const _Float16* p1) {
    h8 lo = *(const h8*)p0;
    h8 hi = *(const h8*)p1;
    return __builtin_shufflevector(lo, hi, 0, 1, 2, 3, 4, 5, 6, 7,
                                           8, 9, 10, 11, 12, 13, 14, 15);
}

__device__ __forceinline__ f8 wmma16(h16 a, h16 b, f8 c) {
    // v_wmma_f32_16x16x32_f16  (neg_a, A, neg_b, B, c_mod, C, reuse_a, reuse_b)
    return __builtin_amdgcn_wmma_f32_16x16x32_f16(false, a, false, b,
                                                  (short)0, c, false, false);
}

// 16-byte global -> LDS copy. CDNA5 async-to-LDS (ASYNCcnt) when declared;
// plain vector copy fallback keeps the source host-pass/portable-safe.
__device__ __forceinline__ void cp16(_Float16* lds, const _Float16* glob) {
#if defined(__gfx1250__) && __has_builtin(__builtin_amdgcn_global_load_async_to_lds_b128)
    __builtin_amdgcn_global_load_async_to_lds_b128(
        (__attribute__((address_space(1))) i4v*)(uintptr_t)glob,
        (__attribute__((address_space(3))) i4v*)(unsigned int)(uintptr_t)lds,
        0, 0);
#else
    *(h8*)lds = *(const h8*)glob;
#endif
}

__device__ __forceinline__ void wait_async() {
#if defined(__gfx1250__) && __has_builtin(__builtin_amdgcn_s_wait_asynccnt)
    __builtin_amdgcn_s_wait_asynccnt(0);
#endif
}

// ---------------------------------------------------------------------------
// Batched strided f16 transpose: dst[z][c][r] = src[zoff(z) + r*rstride + c].
// 64x64 LDS tiles; both global sides coalesced b128, scatter stays in LDS.
// ---------------------------------------------------------------------------
__global__ __launch_bounds__(256)
void transpose_f16(const _Float16* __restrict__ src, _Float16* __restrict__ dst,
                   int R, int C, int rstride, int zdiv, long sb, long sh) {
    __shared__ _Float16 T[64][72];
    const int tid = threadIdx.x;
    const int r0 = blockIdx.y * 64, c0 = blockIdx.x * 64;
    const int z = blockIdx.z;
    const long soff = (long)(z / zdiv) * sb + (long)(z % zdiv) * sh;
    const long doff = (long)z * (long)R * (long)C;
#pragma unroll
    for (int it = 0; it < 2; ++it) {
        int slot = it * 256 + tid;
        int r  = slot >> 3;
        int cc = (slot & 7) * 8;
        *(h8*)&T[r][cc] = *(const h8*)&src[soff + (long)(r0 + r) * rstride + c0 + cc];
    }
    __syncthreads();
#pragma unroll
    for (int it = 0; it < 2; ++it) {
        int slot = it * 256 + tid;
        int c  = slot >> 3;
        int rr = (slot & 7) * 8;
        h8 v;
#pragma unroll
        for (int j = 0; j < 8; ++j) v[j] = T[rr + j][c];
        *(h8*)&dst[doff + (long)(c0 + c) * R + r0 + rr] = v;
    }
}

// ---------------------------------------------------------------------------
// f16 GEMM with pre-transposed B: C[M,N] = A[M,K] * Bt[N,K]^T.
// Block tile 128x128, K-step 32, double-buffered async LDS staging.
// 8 waves: 4 along M (32 rows) x 2 along N (64 cols); 8 WMMA per K-step.
// ---------------------------------------------------------------------------
#define GBM 128
#define GBN 128
#define GBK 32
#define GLD 40   // 32 + 8 halves pad (80B rows, 16B-aligned chunks)

__global__ __launch_bounds__(256)
void gemm_f16_wmma(const _Float16* __restrict__ A, const _Float16* __restrict__ Bt,
                   _Float16* __restrict__ C, int M, int N, int K) {
    __shared__ _Float16 As[2][GBM][GLD];
    __shared__ _Float16 Bs[2][GBN][GLD];

    const int tid  = threadIdx.x;
    const int lane = tid & 31;
    const int wave = tid >> 5;
    const int wm   = wave & 3;
    const int wn   = wave >> 2;
    const int lr   = lane & 15;
    const int lh   = lane >> 4;
    const int m0   = blockIdx.y * GBM;
    const int n0   = blockIdx.x * GBN;

    f8 acc[2][4];
#pragma unroll
    for (int i = 0; i < 2; ++i)
#pragma unroll
        for (int j = 0; j < 4; ++j) acc[i][j] = f8zero();

    auto stage = [&](int buf, int kk) {
#pragma unroll
        for (int it = 0; it < 2; ++it) {
            int slot = it * 256 + tid;
            int row  = slot >> 2;
            int kc   = (slot & 3) * 8;
            cp16(&As[buf][row][kc], &A[(size_t)(m0 + row) * K + kk + kc]);
            cp16(&Bs[buf][row][kc], &Bt[(size_t)(n0 + row) * K + kk + kc]);
        }
        // pull the tile after next toward L2 (global_prefetch_b8)
        if (kk + 2 * GBK < K) {
            __builtin_prefetch(&A[(size_t)(m0 + (tid >> 1)) * K + kk + 2 * GBK], 0, 1);
            __builtin_prefetch(&Bt[(size_t)(n0 + (tid >> 1)) * K + kk + 2 * GBK], 0, 1);
        }
    };

    stage(0, 0);
    wait_async();
    __syncthreads();

    int buf = 0;
    for (int kk = 0; kk < K; kk += GBK) {
        if (kk + GBK < K) stage(buf ^ 1, kk + GBK);   // overlap copy with math

        h16 af[2];
#pragma unroll
        for (int mt = 0; mt < 2; ++mt) {
            int r = wm * 32 + mt * 16 + lr;
            af[mt] = ld_frag(&As[buf][r][16 * lh], &As[buf][r][16 * lh + 8]);
        }
#pragma unroll
        for (int nt = 0; nt < 4; ++nt) {
            int c = wn * 64 + nt * 16 + lr;
            h16 bf = ld_frag(&Bs[buf][c][16 * lh], &Bs[buf][c][16 * lh + 8]);
#pragma unroll
            for (int mt = 0; mt < 2; ++mt)
                acc[mt][nt] = wmma16(af[mt], bf, acc[mt][nt]);
        }

        wait_async();
        __syncthreads();
        buf ^= 1;
    }

    // C/D layout: lane holds col n = lr; VGPR r holds row r + 8*lh.
#pragma unroll
    for (int mt = 0; mt < 2; ++mt)
#pragma unroll
        for (int nt = 0; nt < 4; ++nt)
#pragma unroll
            for (int r = 0; r < 8; ++r) {
                int row = m0 + wm * 32 + mt * 16 + r + 8 * lh;
                int col = n0 + wn * 64 + nt * 16 + lr;
                C[(size_t)row * N + col] = (_Float16)acc[mt][nt][r];
            }
}

// ---------------------------------------------------------------------------
// RoPE in-place. One thread owns the pair (d, d+32) of one head row.
// ---------------------------------------------------------------------------
__global__ __launch_bounds__(256)
void rope_f16(_Float16* __restrict__ t, const _Float16* __restrict__ cosb,
              const _Float16* __restrict__ sinb, int nheads, int total_half) {
    int i = blockIdx.x * blockDim.x + threadIdx.x;
    if (i >= total_half) return;
    int d    = i & 31;
    int base = (i >> 5) * HD;
    int s    = ((base >> 6) / nheads) % Sdim;
    float c0 = (float)cosb[s * HD + d];
    float s0 = (float)sinb[s * HD + d];
    float c1 = (float)cosb[s * HD + d + 32];
    float s1 = (float)sinb[s * HD + d + 32];
    float a  = (float)t[base + d];
    float b  = (float)t[base + d + 32];
    t[base + d]      = (_Float16)(a * c0 - b * s0);
    t[base + d + 32] = (_Float16)(b * c1 + a * s1);
}

// ---------------------------------------------------------------------------
// Flash-attention tile kernel. 128 threads (4 waves), one (b, h, 64-q tile).
// vt is pre-transposed V: vt[(b*NKV+kvh)*HD + d][s]  (key-contiguous rows).
// Online softmax: 2 threads/row, combined with intra-wave __shfl_xor.
// ---------------------------------------------------------------------------
#define QT 64
#define KT 64

__global__ __launch_bounds__(128)
void attn_wmma(const _Float16* __restrict__ q, const _Float16* __restrict__ k,
               const _Float16* __restrict__ vt, _Float16* __restrict__ o) {
    __shared__ _Float16 Qs[QT][80];        // [query][d]  A-layout
    __shared__ _Float16 Ks[KT][80];        // [key][d]    B-layout for QK^T
    __shared__ _Float16 Vs[HD][KT + 8];    // [d][key]    B-layout for P*V
    __shared__ float    Sc[QT][KT + 4];
    __shared__ _Float16 Ps[QT][KT + 8];    // A-layout for P*V
    __shared__ float    m_i[QT], l_i[QT], al[QT];

    const int tid  = threadIdx.x;
    const int lane = tid & 31;
    const int wave = tid >> 5;
    const int lr   = lane & 15;
    const int lh   = lane >> 4;
    const int qt   = blockIdx.x;
    const int h    = blockIdx.y;
    const int b    = blockIdx.z;
    const int q0   = qt * QT;
    const int kvh  = h >> 2;               // N_REP = 4
    const float scale = 0.125f;            // 1/sqrt(64)
    const size_t zoff = (size_t)(b * NKV + kvh) * HD;   // vt row block

    // ---- stage Q tile (async) ----
#pragma unroll
    for (int it = 0; it < (QT * HD) / (128 * 8); ++it) {
        int slot = it * 128 + tid;
        int row  = slot >> 3;
        int dc   = (slot & 7) * 8;
        cp16(&Qs[row][dc],
             &q[((size_t)(b * Sdim + q0 + row) * NH + h) * HD + dc]);
    }
    if (tid < QT) { m_i[tid] = -1e30f; l_i[tid] = 0.f; }

    f8 Oc[4];
#pragma unroll
    for (int t = 0; t < 4; ++t) Oc[t] = f8zero();

    const int nkt = qt + 1;                // causal bound
    for (int kt = 0; kt < nkt; ++kt) {
        const int k0 = kt * KT;
        __syncthreads();                   // prior P*V done before overwrite
        // ---- stage K tile: [key][d], b128 ----
#pragma unroll
        for (int it = 0; it < (KT * HD) / (128 * 8); ++it) {
            int slot = it * 128 + tid;
            int row  = slot >> 3;
            int dc   = (slot & 7) * 8;
            cp16(&Ks[row][dc],
                 &k[((size_t)(b * Sdim + k0 + row) * NKV + kvh) * HD + dc]);
        }
        // ---- stage V^T tile: [d][key], b128 from pre-transposed vt ----
#pragma unroll
        for (int it = 0; it < (HD * KT) / (128 * 8); ++it) {
            int slot = it * 128 + tid;
            int d    = slot >> 3;
            int kc   = (slot & 7) * 8;
            cp16(&Vs[d][kc], &vt[(zoff + d) * Sdim + k0 + kc]);
        }
        wait_async();
        __syncthreads();

        // ---- S = Q K^T : wave owns rows 16w..16w+15; d reduced in 2 steps ----
        f8 st[4];
#pragma unroll
        for (int t = 0; t < 4; ++t) st[t] = f8zero();
#pragma unroll
        for (int step = 0; step < 2; ++step) {
            int mrow = wave * 16 + lr;
            h16 a = ld_frag(&Qs[mrow][32 * step + 16 * lh],
                            &Qs[mrow][32 * step + 16 * lh + 8]);
#pragma unroll
            for (int nt = 0; nt < 4; ++nt) {
                int n = nt * 16 + lr;
                h16 bf = ld_frag(&Ks[n][32 * step + 16 * lh],
                                 &Ks[n][32 * step + 16 * lh + 8]);
                st[nt] = wmma16(a, bf, st[nt]);
            }
        }
#pragma unroll
        for (int nt = 0; nt < 4; ++nt)
#pragma unroll
            for (int r = 0; r < 8; ++r)
                Sc[wave * 16 + r + 8 * lh][nt * 16 + lr] = st[nt][r];
        __syncthreads();

        // ---- online softmax: 2 threads per row, shfl_xor combine ----
        {
            int m    = tid >> 1;
            int half = tid & 1;
            int base = half * 32;
            int qg   = q0 + m;
            float mx = -1e30f;
            for (int ki = base; ki < base + 32; ++ki) {
                float vv = (k0 + ki <= qg) ? Sc[m][ki] * scale : -1e30f;
                Sc[m][ki] = vv;
                mx = fmaxf(mx, vv);
            }
            mx = fmaxf(mx, __shfl_xor(mx, 1, 32));
            float mnew = fmaxf(m_i[m], mx);
            float sum  = 0.f;
            for (int ki = base; ki < base + 32; ++ki) {
                float p = __expf(Sc[m][ki] - mnew);  // masked -> underflow to 0
                Ps[m][ki] = (_Float16)p;
                sum += p;
            }
            sum += __shfl_xor(sum, 1, 32);
            if (half == 0) {
                float a = __expf(m_i[m] - mnew);
                l_i[m]  = l_i[m] * a + sum;
                m_i[m]  = mnew;
                al[m]   = a;
            }
        }
        __syncthreads();

        // ---- rescale O, accumulate O += P * V ----
#pragma unroll
        for (int r = 0; r < 8; ++r) {
            float a = al[wave * 16 + r + 8 * lh];
#pragma unroll
            for (int t = 0; t < 4; ++t) Oc[t][r] *= a;
        }
#pragma unroll
        for (int step = 0; step < 2; ++step) {
            int mrow = wave * 16 + lr;
            h16 a = ld_frag(&Ps[mrow][32 * step + 16 * lh],
                            &Ps[mrow][32 * step + 16 * lh + 8]);
#pragma unroll
            for (int dt = 0; dt < 4; ++dt) {
                int n = dt * 16 + lr;
                h16 bf = ld_frag(&Vs[n][32 * step + 16 * lh],
                                 &Vs[n][32 * step + 16 * lh + 8]);
                Oc[dt] = wmma16(a, bf, Oc[dt]);
            }
        }
    }
    __syncthreads();

    // ---- normalize, store (b, q, h, d) f16 ----
#pragma unroll
    for (int r = 0; r < 8; ++r) {
        int m = wave * 16 + r + 8 * lh;
        float inv = 1.f / l_i[m];
#pragma unroll
        for (int dt = 0; dt < 4; ++dt) {
            int d = dt * 16 + lr;
            o[((size_t)(b * Sdim + q0 + m) * NH + h) * HD + d] =
                (_Float16)(Oc[dt][r] * inv);
        }
    }
}

// ---------------------------------------------------------------------------
// Launcher. Workspace: Q 16M | K 4M | V 4M | attn 16M | Vt 4M | WqT 8M |
// WkT 2M | WvT 2M | WoT 8M  = 64 MB.
// ---------------------------------------------------------------------------
extern "C" void kernel_launch(void* const* d_in, const int* in_sizes, int n_in,
                              void* d_out, int out_size, void* d_ws, size_t ws_size,
                              hipStream_t stream) {
    const _Float16* x    = (const _Float16*)d_in[0];
    const _Float16* wq   = (const _Float16*)d_in[1];
    const _Float16* wk   = (const _Float16*)d_in[2];
    const _Float16* wv   = (const _Float16*)d_in[3];
    const _Float16* wo   = (const _Float16*)d_in[4];
    const _Float16* cosb = (const _Float16*)d_in[5];
    const _Float16* sinb = (const _Float16*)d_in[6];
    _Float16* out = (_Float16*)d_out;

    char* ws = (char*)d_ws;
    const size_t MB = 1024 * 1024;
    _Float16* qb  = (_Float16*)(ws);             // 4096 x 2048
    _Float16* kb  = (_Float16*)(ws + 16 * MB);   // 4096 x 512
    _Float16* vb  = (_Float16*)(ws + 20 * MB);   // 4096 x 512
    _Float16* ab  = (_Float16*)(ws + 24 * MB);   // 4096 x 2048
    _Float16* vtb = (_Float16*)(ws + 40 * MB);   // 16 x 64 x 2048
    _Float16* wqt = (_Float16*)(ws + 44 * MB);   // 2048 x 2048
    _Float16* wkt = (_Float16*)(ws + 52 * MB);   // 512 x 2048
    _Float16* wvt = (_Float16*)(ws + 54 * MB);   // 512 x 2048
    _Float16* wot = (_Float16*)(ws + 56 * MB);   // 2048 x 2048

    const int M = Bdim * Sdim;   // 4096

    // --- one-shot weight transposes: [K][N] -> [N][K] ---
    transpose_f16<<<dim3(Hdim / 64, Hdim / 64, 1), 256, 0, stream>>>(
        wq, wqt, Hdim, Hdim, Hdim, 1, 0, 0);
    transpose_f16<<<dim3((NKV * HD) / 64, Hdim / 64, 1), 256, 0, stream>>>(
        wk, wkt, Hdim, NKV * HD, NKV * HD, 1, 0, 0);
    transpose_f16<<<dim3((NKV * HD) / 64, Hdim / 64, 1), 256, 0, stream>>>(
        wv, wvt, Hdim, NKV * HD, NKV * HD, 1, 0, 0);
    transpose_f16<<<dim3(Hdim / 64, Hdim / 64, 1), 256, 0, stream>>>(
        wo, wot, Hdim, Hdim, Hdim, 1, 0, 0);

    // --- QKV projections (WMMA, double-buffered async staging) ---
    gemm_f16_wmma<<<dim3(Hdim / GBN, M / GBM), 256, 0, stream>>>(
        x, wqt, qb, M, NH * HD, Hdim);
    gemm_f16_wmma<<<dim3((NKV * HD) / GBN, M / GBM), 256, 0, stream>>>(
        x, wkt, kb, M, NKV * HD, Hdim);
    gemm_f16_wmma<<<dim3((NKV * HD) / GBN, M / GBM), 256, 0, stream>>>(
        x, wvt, vb, M, NKV * HD, Hdim);

    // --- RoPE on Q and K ---
    int nq = M * NH * (HD / 2);
    rope_f16<<<(nq + 255) / 256, 256, 0, stream>>>(qb, cosb, sinb, NH, nq);
    int nk = M * NKV * (HD / 2);
    rope_f16<<<(nk + 255) / 256, 256, 0, stream>>>(kb, cosb, sinb, NKV, nk);

    // --- V transpose: (b,s,kvh,d) -> vt[(b*NKV+kvh)*HD + d][s] ---
    transpose_f16<<<dim3(HD / 64, Sdim / 64, Bdim * NKV), 256, 0, stream>>>(
        vb, vtb, Sdim, HD, NKV * HD, NKV,
        (long)Sdim * NKV * HD, (long)HD);

    // --- causal flash attention ---
    attn_wmma<<<dim3(Sdim / QT, NH, Bdim), 128, 0, stream>>>(qb, kb, vtb, ab);

    // --- output projection straight into d_out ---
    gemm_f16_wmma<<<dim3(Hdim / GBN, M / GBM), 256, 0, stream>>>(
        ab, wot, out, M, Hdim, Hdim);
}
